// MultiheadAttention_81346680587096
// MI455X (gfx1250) — compile-verified, matrix operand order checked by reading, not run on hardware
//
#include <hip/hip_runtime.h>

// ---------------------------------------------------------------------------
// MI455X (gfx1250) fused QKV + causal MHA, bf16 WMMA with fp32 accumulation.
// B=4, S=2048, D=1024, H=16, DH=64
// GEMM: 128x128 block tile, K staged 64-wide through LDS, software-pipelined
//       (global prefetch into VGPRs overlapped with the WMMA burst).
// Attention: 128-query block, 32-key tiles staged in LDS shared by 8 waves,
//       same register-pipelined staging.
// ---------------------------------------------------------------------------

typedef __attribute__((ext_vector_type(16))) __bf16        v16bf;
typedef __attribute__((ext_vector_type(8)))  float         v8f;
typedef __attribute__((ext_vector_type(4)))  unsigned int  v4u;

#define BB   4
#define SS   2048
#define DD   1024
#define HH   16
#define DH   64
#define ND   3072          // 3*D
#define MM   8192          // B*S

static __device__ __forceinline__ __bf16 f2bf(float f) { return (__bf16)f; }

// Load one WMMA 16-bit operand fragment (A or B) for this lane.
// Per CDNA5 ISA 16-bit layout: lane l (g=l/16): slots 0-7 hold K=8g..8g+7,
// slots 8-15 hold K=8g+16..8g+23 -> two contiguous 16B loads.
static __device__ __forceinline__ v16bf load_frag(const __bf16* p) {
    union { v4u u[2]; v16bf v; } f;
    f.u[0] = *(const v4u*)(p);
    f.u[1] = *(const v4u*)(p + 16);
    return f.v;
}

static __device__ __forceinline__ v8f wmma_bf16(v16bf a, v16bf b, v8f c) {
    return __builtin_amdgcn_wmma_f32_16x16x32_bf16(
        /*neg_a=*/false, a, /*neg_b=*/false, b,
        /*c_mod=*/(short)0, c, /*reuse_a=*/false, /*reuse_b=*/false);
}

// ---------------------------------------------------------------------------
// Kernel 1: fp32 -> bf16 elementwise convert (for X).
// ---------------------------------------------------------------------------
__global__ void cvt_bf16_kernel(const float* __restrict__ in,
                                __bf16* __restrict__ out, int n) {
    for (int i = blockIdx.x * blockDim.x + threadIdx.x; i < n;
         i += gridDim.x * blockDim.x)
        out[i] = f2bf(in[i]);
}

// ---------------------------------------------------------------------------
// Kernel 2: W[1024][3072] fp32 -> Wt[3072][1024] bf16 (transposed).
// ---------------------------------------------------------------------------
__global__ void cvt_wT_kernel(const float* __restrict__ W,
                              __bf16* __restrict__ Wt) {
    const int total = ND * DD;
    for (int i = blockIdx.x * blockDim.x + threadIdx.x; i < total;
         i += gridDim.x * blockDim.x) {
        int n = i >> 10;        // column of W  (0..3071)
        int k = i & 1023;       // row of W     (0..1023)
        Wt[i] = f2bf(W[(size_t)k * ND + n]);
    }
}

// ---------------------------------------------------------------------------
// Kernel 3: QKV GEMM.  C[8192,3072] = Xb @ Wt^T + bias, scattered into
//   Qb,Kb : [b][h][s][dh] bf16 (dh contiguous)
//   Vt    : [b][h][dh][s] bf16 (s contiguous; pre-transposed for P·V)
// Block tile 128(M) x 128(N), K staged 64-wide through LDS, shared by all
// 8 waves (wave grid 4x2, wave tile 32x64, 16 WMMA per stage per wave).
// Register-pipelined: next stage's global loads issued before the compute
// phase so their latency hides behind the WMMA burst.
// ---------------------------------------------------------------------------
__global__ __launch_bounds__(256) void qkv_gemm_kernel(
    const __bf16* __restrict__ Xb, const __bf16* __restrict__ Wt,
    const float* __restrict__ bias,
    __bf16* __restrict__ Qb, __bf16* __restrict__ Kb, __bf16* __restrict__ Vt) {

    __shared__ __align__(16) __bf16 Ast[128][72];   // 18 KB (stride pad 8)
    __shared__ __align__(16) __bf16 Bst[128][72];   // 18 KB

    const int t    = threadIdx.x;
    const int lane = t & 31;
    const int w    = t >> 5;
    const int g    = lane >> 4;      // half-wave select
    const int mm   = lane & 15;      // row/col within 16
    const int mw   = w & 3;          // wave M index (0..3)
    const int nw   = w >> 2;         // wave N index (0..1)

    const int mBlock = blockIdx.y * 128;
    const int nBlock = blockIdx.x * 128;

    // cooperative-stage indices: thread covers rows ldRow+32i, 16B chunk ldOct
    const int ldRow = t >> 3;
    const int ldOct = (t & 7) * 8;

    v8f acc[2][4];
#pragma unroll
    for (int ms = 0; ms < 2; ++ms)
#pragma unroll
        for (int ns = 0; ns < 4; ++ns)
#pragma unroll
            for (int i = 0; i < 8; ++i) acc[ms][ns][i] = 0.f;

    // ---- pipeline prologue: fetch stage kk=0 into registers ----------------
    v4u aReg[4], bReg[4];
#pragma unroll
    for (int i = 0; i < 4; ++i) {
        const int row = ldRow + 32 * i;
        aReg[i] = *(const v4u*)(Xb + (size_t)(mBlock + row) * DD + ldOct);
        bReg[i] = *(const v4u*)(Wt + (size_t)(nBlock + row) * DD + ldOct);
    }

    for (int kk = 0; kk < DD; kk += 64) {
        __syncthreads();               // previous stage fully consumed
#pragma unroll
        for (int i = 0; i < 4; ++i) {
            const int row = ldRow + 32 * i;
            *(v4u*)&Ast[row][ldOct] = aReg[i];
            *(v4u*)&Bst[row][ldOct] = bReg[i];
        }
        // issue next stage's global loads now; they complete during compute
        if (kk + 64 < DD) {
#pragma unroll
            for (int i = 0; i < 4; ++i) {
                const int row = ldRow + 32 * i;
                aReg[i] = *(const v4u*)(Xb + (size_t)(mBlock + row) * DD + (kk + 64) + ldOct);
                bReg[i] = *(const v4u*)(Wt + (size_t)(nBlock + row) * DD + (kk + 64) + ldOct);
            }
        }
        __syncthreads();               // stores visible to all waves
#pragma unroll
        for (int kc = 0; kc < 64; kc += 32) {
            const v16bf a0 = load_frag(&Ast[mw * 32 + mm][kc + 8 * g]);
            const v16bf a1 = load_frag(&Ast[mw * 32 + 16 + mm][kc + 8 * g]);
#pragma unroll
            for (int ns = 0; ns < 4; ++ns) {
                const v16bf bf = load_frag(&Bst[nw * 64 + ns * 16 + mm][kc + 8 * g]);
                acc[0][ns] = wmma_bf16(a0, bf, acc[0][ns]);
                acc[1][ns] = wmma_bf16(a1, bf, acc[1][ns]);
            }
        }
    }

    // epilogue: bias add, bf16 convert, scatter into Q/K/Vt layouts
#pragma unroll
    for (int ns = 0; ns < 4; ++ns) {
        const int col = nBlock + nw * 64 + ns * 16 + mm;   // constant per lane
        const float bv = bias[col];
        const int hh  = col / 192;                          // head
        const int rem = col - hh * 192;
        const int rr  = rem >> 6;                           // 0=q 1=k 2=v
        const int dh  = rem & 63;
#pragma unroll
        for (int ms = 0; ms < 2; ++ms) {
#pragma unroll
            for (int r = 0; r < 8; ++r) {
                const int grow = mBlock + mw * 32 + ms * 16 + r + 8 * g;
                const int bb   = grow >> 11;
                const int sIdx = grow & 2047;
                const __bf16 val = f2bf(acc[ms][ns][r] + bv);
                if (rr == 0)
                    Qb[(((size_t)bb * HH + hh) * SS + sIdx) * DH + dh] = val;
                else if (rr == 1)
                    Kb[(((size_t)bb * HH + hh) * SS + sIdx) * DH + dh] = val;
                else
                    Vt[(((size_t)bb * HH + hh) * DH + dh) * SS + sIdx] = val;
            }
        }
    }
}

// ---------------------------------------------------------------------------
// Kernel 4: flash attention.  grid = (S/128, B*H), block = 256 (8 waves).
// All 8 waves iterate the same 32-key tiles; K (32x64) and V^T (64x32) are
// cooperatively staged in LDS once per tile, register-pipelined so the next
// tile's global loads overlap the current tile's compute.  Waves past their
// causal range skip compute (wave-uniform branch; EXEC all ones in WMMA).
// ---------------------------------------------------------------------------
__global__ __launch_bounds__(256) void attention_kernel(
    const __bf16* __restrict__ Qb, const __bf16* __restrict__ Kb,
    const __bf16* __restrict__ Vt, float* __restrict__ out) {

    __shared__ __align__(16) __bf16 Kst[32][72];      // keys x dh (pad 8)
    __shared__ __align__(16) __bf16 Vst[64][40];      // dh x keys (pad 8)
    __shared__ __align__(16) __bf16 Pst[8][16][32];   // per-wave P staging

    const int t    = threadIdx.x;
    const int lane = t & 31;
    const int w    = t >> 5;
    const int g    = lane >> 4;
    const int mmL  = lane & 15;

    const int Q0 = blockIdx.x * 128;
    const int q0 = Q0 + w * 16;
    const int bh = blockIdx.y;
    const int b  = bh >> 4;
    const int h  = bh & 15;

    const __bf16* Qbh = Qb + (size_t)bh * SS * DH;
    const __bf16* Kbh = Kb + (size_t)bh * SS * DH;
    const __bf16* Vbh = Vt + (size_t)bh * DH * SS;

    // Q fragments (resident across the whole key loop): d 0..31 and 32..63
    const __bf16* pq = Qbh + (size_t)(q0 + mmL) * DH + 8 * g;
    const v16bf qf0 = load_frag(pq);
    const v16bf qf1 = load_frag(pq + 32);

    float mI[8], lI[8];
    v8f O[4];
#pragma unroll
    for (int r = 0; r < 8; ++r) { mI[r] = -1e30f; lI[r] = 0.f; }
#pragma unroll
    for (int tt = 0; tt < 4; ++tt)
#pragma unroll
        for (int i = 0; i < 8; ++i) O[tt][i] = 0.f;

    const float scale = 0.125f;              // 1/sqrt(64)

    // cooperative-stage indices (constant per thread)
    const int krow = t >> 3, koct = (t & 7) * 8;    // K: 32 rows x 64
    const int vrow = t >> 2, voct = (t & 3) * 8;    // V: 64 rows x 32

    const int jtEnd = (Q0 + 127) / 32;              // block-wide causal bound

    // ---- pipeline prologue: fetch tile jt=0 into registers -----------------
    v4u kReg = *(const v4u*)(Kbh + (size_t)krow * DH + koct);
    v4u vReg = *(const v4u*)(Vbh + (size_t)vrow * SS + voct);

    for (int jt = 0; jt <= jtEnd; ++jt) {
        const int jb = jt * 32;

        __syncthreads();   // previous tile fully consumed before overwrite
        *(v4u*)&Kst[krow][koct] = kReg;
        *(v4u*)&Vst[vrow][voct] = vReg;
        if (jt < jtEnd) {  // prefetch next tile; completes during compute
            kReg = *(const v4u*)(Kbh + (size_t)(jb + 32 + krow) * DH + koct);
            vReg = *(const v4u*)(Vbh + (size_t)vrow * SS + (jb + 32) + voct);
        }
        __syncthreads();

        if (jb <= q0 + 15) {                 // wave-uniform causal skip
            const bool needMask = (jb + 31) > q0;

            // ---- scores: 2 x 16x16 tiles over 32 keys, K=64 in 2 steps ----
            v8f s0, s1;
#pragma unroll
            for (int i = 0; i < 8; ++i) { s0[i] = 0.f; s1[i] = 0.f; }
            s0 = wmma_bf16(qf0, load_frag(&Kst[mmL][8 * g]),            s0);
            s0 = wmma_bf16(qf1, load_frag(&Kst[mmL][32 + 8 * g]),       s0);
            s1 = wmma_bf16(qf0, load_frag(&Kst[16 + mmL][8 * g]),       s1);
            s1 = wmma_bf16(qf1, load_frag(&Kst[16 + mmL][32 + 8 * g]),  s1);

            // ---- online softmax (row = r + 8g over 16 lanes) --------------
#pragma unroll
            for (int r = 0; r < 8; ++r) {
                float a = s0[r] * scale;
                float c = s1[r] * scale;
                if (needMask) {
                    const int qi = q0 + r + 8 * g;
                    if (jb + mmL      > qi) a = -1e30f;
                    if (jb + 16 + mmL > qi) c = -1e30f;
                }
                float tm = fmaxf(a, c);
#pragma unroll
                for (int off = 1; off < 16; off <<= 1)
                    tm = fmaxf(tm, __shfl_xor(tm, off, 32));
                const float nm = fmaxf(mI[r], tm);
                const float al = __expf(mI[r] - nm);
                const float p0 = __expf(a - nm);
                const float p1 = __expf(c - nm);
                float rs = p0 + p1;
#pragma unroll
                for (int off = 1; off < 16; off <<= 1)
                    rs += __shfl_xor(rs, off, 32);
                lI[r] = lI[r] * al + rs;
                mI[r] = nm;
#pragma unroll
                for (int tt = 0; tt < 4; ++tt) O[tt][r] *= al;
                Pst[w][r + 8 * g][mmL]      = f2bf(p0);
                Pst[w][r + 8 * g][mmL + 16] = f2bf(p1);
            }

            // ---- P·V: A-frag of P from LDS, B-frags from staged V^T -------
            const v16bf pf = load_frag(&Pst[w][mmL][8 * g]);
#pragma unroll
            for (int tt = 0; tt < 4; ++tt)
                O[tt] = wmma_bf16(pf, load_frag(&Vst[tt * 16 + mmL][8 * g]), O[tt]);
        }
    }

    // ---- normalize and write out [b][s][h*64+dh] fp32 ----------------------
#pragma unroll
    for (int r = 0; r < 8; ++r) {
        const float inv = 1.0f / lI[r];
        const int row = q0 + r + 8 * g;
        float* po = out + ((size_t)b * SS + row) * DD + h * DH;
#pragma unroll
        for (int tt = 0; tt < 4; ++tt)
            po[tt * 16 + mmL] = O[tt][r] * inv;
    }
}

// ---------------------------------------------------------------------------
// Launch
// ---------------------------------------------------------------------------
extern "C" void kernel_launch(void* const* d_in, const int* in_sizes, int n_in,
                              void* d_out, int out_size, void* d_ws, size_t ws_size,
                              hipStream_t stream) {
    const float* x    = (const float*)d_in[0];   // [B,S,D]
    const float* Wq   = (const float*)d_in[1];   // [D,3D]
    const float* bq   = (const float*)d_in[2];   // [3D]
    float* out        = (float*)d_out;           // [B,S,D]

    char* ws = (char*)d_ws;
    const size_t xbBytes = (size_t)MM * DD * 2;           // 16 MiB
    const size_t wtBytes = (size_t)ND * DD * 2;           //  6 MiB
    const size_t qkvEach = (size_t)BB * HH * SS * DH * 2; // 16 MiB
    __bf16* Xb = (__bf16*)(ws);
    __bf16* Wt = (__bf16*)(ws + xbBytes);
    __bf16* Qb = (__bf16*)(ws + xbBytes + wtBytes);
    __bf16* Kb = (__bf16*)(ws + xbBytes + wtBytes + qkvEach);
    __bf16* Vt = (__bf16*)(ws + xbBytes + wtBytes + 2 * qkvEach);

    cvt_bf16_kernel<<<2048, 256, 0, stream>>>(x, Xb, MM * DD);
    cvt_wT_kernel<<<2048, 256, 0, stream>>>(Wq, Wt);
    qkv_gemm_kernel<<<dim3(ND / 128, MM / 128), 256, 0, stream>>>(Xb, Wt, bq, Qb, Kb, Vt);
    attention_kernel<<<dim3(SS / 128, BB * HH), 256, 0, stream>>>(Qb, Kb, Vt, out);
}
